// PerformerViTSTRING_13700945674725
// MI455X (gfx1250) — compile-verified, matrix operand order checked by reading, not run on hardware
//
#include <hip/hip_runtime.h>
#include <math.h>

// Problem constants (from reference)
#define B_   32
#define E_   768
#define H_   12
#define DH_  64
#define M_   256
#define L_   6
#define P_   16
#define GS_  14
#define NC_  1000
#define N_   (GS_*GS_)     // 196
#define S_   (N_+1)        // 197
#define MLP_ (4*E_)        // 3072
#define NF_  (E_/2+1)      // 385 rfft bins

#define LDSROW   40                // ushorts per LDS row (80 B: 16B-mult, 16-bank stride)
#define BUFELEMS (64*LDSROW)       // one tile buffer: 2560 ushorts = 5120 B

typedef __attribute__((ext_vector_type(16))) __bf16 v16bf;
typedef __attribute__((ext_vector_type(8)))  float  v8f;

__device__ __forceinline__ unsigned short f2bf(float f) {
  unsigned u = __float_as_uint(f);
  u += 0x7fffu + ((u >> 16) & 1u);       // round-to-nearest-even truncation
  return (unsigned short)(u >> 16);
}

__device__ __forceinline__ unsigned lds_off(const void* p) {
  // flat shared pointer carries the LDS byte offset in its low 32 bits
  return (unsigned)(size_t)p;
}

// fp32 -> bf16 weight pre-conversion (runs once per launch; weights then L2-resident)
__global__ void cvt_bf16(const float* __restrict__ src,
                         unsigned short* __restrict__ dst, int n) {
  int i = blockIdx.x * blockDim.x + threadIdx.x;
  if (i < n) dst[i] = f2bf(src[i]);
}

// ---------------------------------------------------------------------------
// WMMA GEMM: C[R,Cc] = A[R,K](f32) @ Bw[Cc,K](bf16)^T (+bias)(+resid)(+gelu)
// Block tile 64x64, 256 thr = 8 waves (2x4); each wave: 2 WMMA tiles (32x16).
// Double-buffered LDS, software pipelined:
//   - B tile DMA'd by global_load_async_to_lds_b128 into alternate buffer,
//     issued right after the barrier so the DMA overlaps the WMMAs.
//   - A tile prefetched to registers, converted f32->bf16, stored b128.
//   - one s_wait_asynccnt + one barrier per K-step.
// Row indices CLAMPED (not predicated) -> branchless b128 staging.
// ---------------------------------------------------------------------------
__global__ __launch_bounds__(256)
void gemm_wmma(const float* __restrict__ A, const unsigned short* __restrict__ Bw,
               const float* __restrict__ bias, const float* __restrict__ resid,
               float* __restrict__ C, int R, int Cc, int K, int gelu)
{
  __shared__ __align__(16) unsigned short As[2*BUFELEMS];
  __shared__ __align__(16) unsigned short Bs[2*BUFELEMS];
  const int tid  = threadIdx.x;
  const int lane = tid & 31, wave = tid >> 5;
  const int wm = wave >> 2, wn = wave & 3;            // 2 x 4 wave tiles
  const int rowBase = blockIdx.y * 64, colBase = blockIdx.x * 64;
  const int half = lane >> 4, fr = lane & 15;

  // staging mapping: thread -> (row 0..63, 16B chunk 0..3)
  const int sr = tid >> 2, sq = tid & 3;
  int ar = rowBase + sr; if (ar >= R)  ar = R - 1;    // clamp (stores are guarded)
  int br = colBase + sr; if (br >= Cc) br = Cc - 1;
  const float*          Arow = A  + (size_t)ar * K + sq * 8;
  const unsigned short* Brow = Bw + (size_t)br * K + sq * 8;
  const unsigned ldsB0 = lds_off(&Bs[0]) + (unsigned)(sr * 80 + sq * 16);
  unsigned short* AsDst = &As[sr * LDSROW + sq * 8];

  v8f acc0 = {0.f,0.f,0.f,0.f,0.f,0.f,0.f,0.f};
  v8f acc1 = {0.f,0.f,0.f,0.f,0.f,0.f,0.f,0.f};

  const int T = K >> 5;                               // 32-wide K tiles

  // ---- prologue: stage tile 0 into buffer 0 ----
  asm volatile("global_load_async_to_lds_b128 %0, %1, off"
               :: "v"(ldsB0), "v"(Brow) : "memory");
  {
    const float4* ap = reinterpret_cast<const float4*>(Arow);
    float4 a0 = ap[0], a1 = ap[1];
    unsigned p0 = ((unsigned)f2bf(a0.y) << 16) | f2bf(a0.x);
    unsigned p1 = ((unsigned)f2bf(a0.w) << 16) | f2bf(a0.z);
    unsigned p2 = ((unsigned)f2bf(a1.y) << 16) | f2bf(a1.x);
    unsigned p3 = ((unsigned)f2bf(a1.w) << 16) | f2bf(a1.z);
    *reinterpret_cast<uint4*>(AsDst) = make_uint4(p0, p1, p2, p3);
  }

  for (int i = 0; i < T; ++i) {
    const int  b    = i & 1;
    const bool more = (i + 1 < T);
    float4 n0, n1;
    if (more) {                                       // A prefetch to registers
      const float4* ap = reinterpret_cast<const float4*>(Arow + (size_t)(i+1)*32);
      n0 = ap[0]; n1 = ap[1];
    }
    asm volatile("s_wait_asynccnt 0x0" ::: "memory"); // tile i's DMA landed
    __syncthreads();                                  // all waves staged tile i;
                                                      // all reads of buf b^1 done
    if (more) {
      // issue next B DMA now -> overlaps with the WMMAs below
      asm volatile("global_load_async_to_lds_b128 %0, %1, off"
                   :: "v"(ldsB0 + (unsigned)((b ^ 1) * (BUFELEMS * 2))),
                      "v"(Brow + (size_t)(i+1)*32) : "memory");
      unsigned p0 = ((unsigned)f2bf(n0.y) << 16) | f2bf(n0.x);
      unsigned p1 = ((unsigned)f2bf(n0.w) << 16) | f2bf(n0.z);
      unsigned p2 = ((unsigned)f2bf(n1.y) << 16) | f2bf(n1.x);
      unsigned p3 = ((unsigned)f2bf(n1.w) << 16) | f2bf(n1.z);
      *reinterpret_cast<uint4*>(AsDst + (b ^ 1) * BUFELEMS) = make_uint4(p0, p1, p2, p3);
    }

    const unsigned short* Asb = &As[b * BUFELEMS];
    const unsigned short* Bsb = &Bs[b * BUFELEMS];
    // Fragments per ISA 16-bit A 16x32 layout: VGPR v <-> K = 16*(v>=4)+8*half+2*(v&3)
    union { v16bf v; unsigned u[8]; } af0, af1, bfm;
#pragma unroll
    for (int vv = 0; vv < 8; ++vv) {
      int kb = ((vv >> 2) << 4) + (half << 3) + ((vv & 3) << 1);
      af0.u[vv] = *reinterpret_cast<const unsigned*>(&Asb[(wm*32 +      fr)*LDSROW + kb]);
      af1.u[vv] = *reinterpret_cast<const unsigned*>(&Asb[(wm*32 + 16 + fr)*LDSROW + kb]);
      bfm.u[vv] = *reinterpret_cast<const unsigned*>(&Bsb[(wn*16 +      fr)*LDSROW + kb]);
    }
    acc0 = __builtin_amdgcn_wmma_f32_16x16x32_bf16(false, af0.v, false, bfm.v,
                                                   (short)0, acc0, false, false);
    acc1 = __builtin_amdgcn_wmma_f32_16x16x32_bf16(false, af1.v, false, bfm.v,
                                                   (short)0, acc1, false, false);
  }

  // ---- epilogue (C/D layout: VGPR r -> M = r + 8*half ; N = lane&15) ----
  const int n = colBase + wn*16 + fr;
  if (n >= Cc) return;
  const float bv = bias ? bias[n] : 0.f;
  const int  m0base = rowBase + wm*32 + 8*half;
  const bool full   = (rowBase + 64 <= R);

  float o0[8], o1[8];
#pragma unroll
  for (int r = 0; r < 8; ++r) { o0[r] = acc0[r] + bv; o1[r] = acc1[r] + bv; }
  if (resid) {
#pragma unroll
    for (int r = 0; r < 8; ++r) {
      int m0 = m0base + r, m1 = m0 + 16;
      if (full || m0 < R) o0[r] += resid[(size_t)m0*Cc + n];
      if (full || m1 < R) o1[r] += resid[(size_t)m1*Cc + n];
    }
  }
  if (gelu) {
#pragma unroll
    for (int r = 0; r < 8; ++r) {
      o0[r] = 0.5f * o0[r] * (1.f + erff(o0[r] * 0.70710678118f));
      o1[r] = 0.5f * o1[r] * (1.f + erff(o1[r] * 0.70710678118f));
    }
  }
#pragma unroll
  for (int r = 0; r < 8; ++r) {
    int m0 = m0base + r, m1 = m0 + 16;
    if (full || m0 < R) C[(size_t)m0*Cc + n] = o0[r];
    if (full || m1 < R) C[(size_t)m1*Cc + n] = o1[r];
  }
}

// ---------------------------------------------------------------------------
// Patch gather: x(B,3,224,224) -> xp(B,N,768) with c = ch*256 + py*16 + px
// ---------------------------------------------------------------------------
__global__ void gather_patches(const float* __restrict__ x, float* __restrict__ xp,
                               int total) {
  int idx = blockIdx.x * blockDim.x + threadIdx.x;
  if (idx >= total) return;
  int c = idx % 768;
  int n = (idx / 768) % N_;
  int b = idx / (768 * N_);
  int ch = c >> 8, py = (c >> 4) & 15, px = c & 15;
  int gy = n / GS_, gx = n % GS_;
  xp[idx] = x[(((size_t)b*3 + ch)*224 + gy*P_ + py)*224 + gx*P_ + px];
}

// ---------------------------------------------------------------------------
// STRING: cimag[k][f] = Im(rfft(circ_params[k]))[f] = -sum_e c[e] sin(2pi f e/E)
// ---------------------------------------------------------------------------
__global__ void dft_imag(const float* __restrict__ circ, float* __restrict__ cim) {
  int k = blockIdx.x, f = threadIdx.x;
  if (f >= NF_) return;
  const float* cp = circ + k*E_;
  float th = 6.283185307179586f * (float)f / (float)E_;
  float ss, cs; __sincosf(th, &ss, &cs);
  float c = 1.f, s = 0.f, acc = 0.f;
  for (int e = 0; e < E_; ++e) {
    acc += cp[e] * s;
    float cn = c*cs - s*ss; s = s*cs + c*ss; c = cn;
  }
  cim[k*NF_ + f] = -acc;
}

// One STRING pass over coord dim `kdim`: dst = irfft( exp(i*2*pos*cimag) * rfft(src) )
__global__ __launch_bounds__(256)
void string_pass(const float* __restrict__ src, float* __restrict__ dst,
                 const int* __restrict__ pos, const float* __restrict__ cim, int kdim)
{
  __shared__ float xr[E_];
  __shared__ float yr[NF_], yi[NF_];
  int bn = blockIdx.x;                    // b*N + n
  int n  = bn % N_;
  int tid = threadIdx.x;
  const float* row = src + (size_t)bn * E_;
  for (int e = tid; e < E_; e += 256) xr[e] = row[e];
  __syncthreads();
  float p = (float)pos[n*2 + kdim];
  for (int f = tid; f < NF_; f += 256) {
    float th = -6.283185307179586f * (float)f / (float)E_;
    float ss, cs; __sincosf(th, &ss, &cs);
    float c = 1.f, s = 0.f, ar = 0.f, ai = 0.f;
    for (int e = 0; e < E_; ++e) {       // forward DFT via rotation recurrence
      float v = xr[e];
      ar += v * c; ai += v * s;
      float cn = c*cs - s*ss; s = s*cs + c*ss; c = cn;
    }
    float u = 2.f * p * cim[kdim*NF_ + f];
    float su, cu; __sincosf(u, &su, &cu);
    yr[f] = ar*cu - ai*su;
    yi[f] = ar*su + ai*cu;
  }
  __syncthreads();
  float* out = dst + (size_t)bn * E_;
  for (int e = tid; e < E_; e += 256) {
    float ph = 6.283185307179586f * (float)e / (float)E_;
    float ss, cs; __sincosf(ph, &ss, &cs);
    float c = cs, s = ss;                 // f = 1
    float acc = yr[0] + ((e & 1) ? -yr[NF_-1] : yr[NF_-1]);   // f=0, Nyquist
    float sum = 0.f;
    for (int f = 1; f < NF_-1; ++f) {
      sum += yr[f]*c - yi[f]*s;
      float cn = c*cs - s*ss; s = s*cs + c*ss; c = cn;
    }
    out[e] = (acc + 2.f*sum) / (float)E_;
  }
}

__global__ void concat_cls(const float* __restrict__ strd, const float* __restrict__ cls_t,
                           const float* __restrict__ cls_p, float* __restrict__ h, int total) {
  int idx = blockIdx.x * blockDim.x + threadIdx.x;
  if (idx >= total) return;
  int e = idx % E_;
  int s = (idx / E_) % S_;
  int b = idx / (E_ * S_);
  h[idx] = (s == 0) ? (cls_t[e] + cls_p[e])
                    : strd[((size_t)b*N_ + (s-1))*E_ + e];
}

// ---------------------------------------------------------------------------
// LayerNorm over E per row
// ---------------------------------------------------------------------------
__global__ __launch_bounds__(256)
void layernorm(const float* __restrict__ x, float* __restrict__ y,
               const float* __restrict__ g, const float* __restrict__ bta, int rows) {
  __shared__ float red[256];
  int row = blockIdx.x; if (row >= rows) return;
  int tid = threadIdx.x;
  const float* xr = x + (size_t)row*E_;
  float s = 0.f;
  for (int e = tid; e < E_; e += 256) s += xr[e];
  red[tid] = s; __syncthreads();
  for (int off = 128; off > 0; off >>= 1) { if (tid < off) red[tid] += red[tid+off]; __syncthreads(); }
  float mu = red[0] / (float)E_;
  __syncthreads();
  float v = 0.f;
  for (int e = tid; e < E_; e += 256) { float d = xr[e] - mu; v += d*d; }
  red[tid] = v; __syncthreads();
  for (int off = 128; off > 0; off >>= 1) { if (tid < off) red[tid] += red[tid+off]; __syncthreads(); }
  float inv = rsqrtf(red[0] / (float)E_ + 1e-5f);
  for (int e = tid; e < E_; e += 256)
    y[(size_t)row*E_ + e] = (xr[e] - mu) * inv * g[e] + bta[e];
}

// ---------------------------------------------------------------------------
// FAVOR+ feature maps
// ---------------------------------------------------------------------------
__device__ __forceinline__ unsigned fenc(float f) {
  unsigned u = __float_as_uint(f);
  return (u & 0x80000000u) ? ~u : (u | 0x80000000u);
}
__device__ __forceinline__ float fdec(unsigned k) {
  unsigned u = (k & 0x80000000u) ? (k & 0x7fffffffu) : ~k;
  return __uint_as_float(u);
}

__global__ void init_gmax(unsigned* g) { *g = fenc(-3.0e38f); }

__global__ __launch_bounds__(256)
void phi_q(const float* __restrict__ qkv, const float* __restrict__ rf,
           float* __restrict__ qp) {
  __shared__ float qs[DH_];
  __shared__ float red[256];
  int bhs = blockIdx.x;
  int s  = bhs % S_;
  int hh = (bhs / S_) % H_;
  int b  = bhs / (S_ * H_);
  int tid = threadIdx.x;
  const float scale = 0.35355339059327373f;          // 64^-0.25
  if (tid < DH_) qs[tid] = qkv[(size_t)(b*S_+s)*(3*E_) + hh*DH_ + tid] * scale;
  __syncthreads();
  float qq = (tid < DH_) ? qs[tid]*qs[tid] : 0.f;
  red[tid] = qq; __syncthreads();
  for (int off = 128; off > 0; off >>= 1) { if (tid < off) red[tid] += red[tid+off]; __syncthreads(); }
  float hq = 0.5f * red[0];
  __syncthreads();
  float pr = 0.f;
  const float* rfm = rf + tid*DH_;
#pragma unroll 8
  for (int d = 0; d < DH_; ++d) pr += qs[d]*rfm[d];
  float logit = pr - hq;
  red[tid] = logit; __syncthreads();
  for (int off = 128; off > 0; off >>= 1) { if (tid < off) red[tid] = fmaxf(red[tid], red[tid+off]); __syncthreads(); }
  float mx = red[0];
  qp[((size_t)((b*H_+hh)*S_ + s))*M_ + tid] = expf(logit - mx) * 0.0625f;  // M^-0.5
}

__global__ __launch_bounds__(256)
void phi_k_logits(const float* __restrict__ qkv, const float* __restrict__ rf,
                  float* __restrict__ kl, unsigned* __restrict__ gmax) {
  __shared__ float qs[DH_];
  __shared__ float red[256];
  int bhs = blockIdx.x;
  int s  = bhs % S_;
  int hh = (bhs / S_) % H_;
  int b  = bhs / (S_ * H_);
  int tid = threadIdx.x;
  const float scale = 0.35355339059327373f;
  if (tid < DH_) qs[tid] = qkv[(size_t)(b*S_+s)*(3*E_) + E_ + hh*DH_ + tid] * scale;
  __syncthreads();
  float qq = (tid < DH_) ? qs[tid]*qs[tid] : 0.f;
  red[tid] = qq; __syncthreads();
  for (int off = 128; off > 0; off >>= 1) { if (tid < off) red[tid] += red[tid+off]; __syncthreads(); }
  float hq = 0.5f * red[0];
  __syncthreads();
  float pr = 0.f;
  const float* rfm = rf + tid*DH_;
#pragma unroll 8
  for (int d = 0; d < DH_; ++d) pr += qs[d]*rfm[d];
  float logit = pr - hq;
  kl[((size_t)((b*H_+hh)*S_ + s))*M_ + tid] = logit;
  red[tid] = logit; __syncthreads();
  for (int off = 128; off > 0; off >>= 1) { if (tid < off) red[tid] = fmaxf(red[tid], red[tid+off]); __syncthreads(); }
  if (tid == 0) atomicMax(gmax, fenc(red[0]));
}

__global__ void phi_k_exp(float* __restrict__ kl, const unsigned* __restrict__ gmax,
                          int total) {
  int idx = blockIdx.x * blockDim.x + threadIdx.x;
  if (idx >= total) return;
  float mx = fdec(*gmax);
  kl[idx] = expf(kl[idx] - mx) * 0.0625f;
}

// kv[b,h,m,d] = sum_s kp[b,h,s,m]*v[b,h,s,d]; ksum[b,h,m] = sum_s kp
__global__ __launch_bounds__(256)
void kv_ksum(const float* __restrict__ kp, const float* __restrict__ qkv,
             float* __restrict__ kv, float* __restrict__ ksum) {
  int bh = blockIdx.x;
  int b = bh / H_, hh = bh % H_;
  int tid = threadIdx.x;
  const float* kpb = kp  + (size_t)bh * S_ * M_;
  const float* vb  = qkv + (size_t)b * S_ * 3*E_ + 2*E_ + hh*DH_;
  for (int i = tid; i < M_*DH_; i += 256) {
    int m = i >> 6, d = i & 63;
    float acc = 0.f;
    for (int s = 0; s < S_; ++s)
      acc += kpb[s*M_ + m] * vb[(size_t)s*3*E_ + d];
    kv[(size_t)bh*M_*DH_ + i] = acc;
  }
  {
    int m = tid;                           // blockDim == M_
    float acc = 0.f;
    for (int s = 0; s < S_; ++s) acc += kpb[s*M_ + m];
    ksum[(size_t)bh*M_ + m] = acc;
  }
}

// attn[b,s,h*DH+d] = (sum_m qp*kv) / (sum_m qp*ksum + 1e-6)
__global__ __launch_bounds__(256)
void num_den(const float* __restrict__ qp, const float* __restrict__ kv,
             const float* __restrict__ ksum, float* __restrict__ attn) {
  __shared__ float qpr[M_];
  __shared__ float red[256];
  int bhs = blockIdx.x;
  int s  = bhs % S_;
  int hh = (bhs / S_) % H_;
  int b  = bhs / (S_ * H_);
  int bh = b*H_ + hh;
  int tid = threadIdx.x;
  float q = qp[((size_t)bh*S_ + s)*M_ + tid];
  qpr[tid] = q;
  red[tid] = q * ksum[(size_t)bh*M_ + tid];
  __syncthreads();
  for (int off = 128; off > 0; off >>= 1) { if (tid < off) red[tid] += red[tid+off]; __syncthreads(); }
  float den = red[0] + 1e-6f;
  if (tid < DH_) {
    const float* kvb = kv + (size_t)bh*M_*DH_ + tid;
    float acc = 0.f;
#pragma unroll 4
    for (int m = 0; m < M_; ++m) acc += qpr[m] * kvb[m*DH_];
    attn[(size_t)(b*S_+s)*E_ + hh*DH_ + tid] = acc / den;
  }
}

__global__ void extract_cls(const float* __restrict__ hn, float* __restrict__ cls,
                            int total) {
  int idx = blockIdx.x * blockDim.x + threadIdx.x;
  if (idx >= total) return;
  int b = idx / E_, e = idx % E_;
  cls[idx] = hn[(size_t)(b*S_)*E_ + e];
}

// ---------------------------------------------------------------------------
extern "C" void kernel_launch(void* const* d_in, const int* in_sizes, int n_in,
                              void* d_out, int out_size, void* d_ws, size_t ws_size,
                              hipStream_t stream)
{
  const float* x        = (const float*)d_in[0];
  const float* patch_w  = (const float*)d_in[1];
  const float* patch_b  = (const float*)d_in[2];
  const float* circ     = (const float*)d_in[3];
  const float* cls_t    = (const float*)d_in[4];
  const float* cls_p    = (const float*)d_in[5];
  const float* ln1_g    = (const float*)d_in[6];
  const float* ln1_b    = (const float*)d_in[7];
  const float* wqkv     = (const float*)d_in[8];
  const float* bqkv     = (const float*)d_in[9];
  const float* wo       = (const float*)d_in[10];
  const float* bo       = (const float*)d_in[11];
  const float* ln2_g    = (const float*)d_in[12];
  const float* ln2_b    = (const float*)d_in[13];
  const float* w1       = (const float*)d_in[14];
  const float* b1       = (const float*)d_in[15];
  const float* w2       = (const float*)d_in[16];
  const float* b2       = (const float*)d_in[17];
  const float* rf       = (const float*)d_in[18];
  const float* norm_g   = (const float*)d_in[19];
  const float* norm_b   = (const float*)d_in[20];
  const float* head_w   = (const float*)d_in[21];
  const float* head_b   = (const float*)d_in[22];
  const int*   positions= (const int*)d_in[23];
  (void)in_sizes; (void)n_in; (void)out_size; (void)ws_size;

  float* ws = (float*)d_ws;
  size_t o = 0;
  float* f_h   = ws + o; o += (size_t)B_*S_*E_;        // residual stream
  float* f_a   = ws + o; o += (size_t)B_*S_*E_;        // LN out / attn out
  float* f_qkv = ws + o; o += (size_t)B_*S_*3*E_;      // also xp at start
  float* f_qp  = ws + o; o += (size_t)B_*H_*S_*M_;     // also STRING ping-pong
  float* f_kp  = ws + o; o += (size_t)B_*H_*S_*M_;     // also MLP hidden (same size)
  float* f_kv  = ws + o; o += (size_t)B_*H_*M_*DH_;
  float* f_ks  = ws + o; o += (size_t)B_*H_*M_;
  float* f_cim = ws + o; o += 2*NF_;
  float* f_cls = ws + o; o += (size_t)B_*E_;
  unsigned* f_gmax = (unsigned*)(ws + o); o += 1;
  o = (o + 7) & ~(size_t)7;                            // 16B-align bf16 region
  unsigned short* wbf = (unsigned short*)(ws + o);
  // bf16 weight copies (element offsets)
  const size_t np  = (size_t)E_*768;                   // patch_w
  const size_t nq  = (size_t)L_*3*E_*E_;               // wqkv
  const size_t nwo = (size_t)L_*E_*E_;                 // wo
  const size_t n1  = (size_t)L_*MLP_*E_;               // w1
  const size_t n2  = (size_t)L_*E_*MLP_;               // w2
  const size_t nh  = (size_t)NC_*E_;                   // head_w
  unsigned short* bf_patch = wbf;
  unsigned short* bf_qkv   = bf_patch + np;
  unsigned short* bf_wo    = bf_qkv   + nq;
  unsigned short* bf_w1    = bf_wo    + nwo;
  unsigned short* bf_w2    = bf_w1    + n1;
  unsigned short* bf_head  = bf_w2    + n2;
  float* f_xp  = f_qkv;
  float* f_tmp = f_qp;

  dim3 blk(256);

  // Weight conversion (once per launch)
  cvt_bf16<<<(int)((np +255)/256), blk, 0, stream>>>(patch_w, bf_patch, (int)np);
  cvt_bf16<<<(int)((nq +255)/256), blk, 0, stream>>>(wqkv,    bf_qkv,   (int)nq);
  cvt_bf16<<<(int)((nwo+255)/256), blk, 0, stream>>>(wo,      bf_wo,    (int)nwo);
  cvt_bf16<<<(int)((n1 +255)/256), blk, 0, stream>>>(w1,      bf_w1,    (int)n1);
  cvt_bf16<<<(int)((n2 +255)/256), blk, 0, stream>>>(w2,      bf_w2,    (int)n2);
  cvt_bf16<<<(int)((nh +255)/256), blk, 0, stream>>>(head_w,  bf_head,  (int)nh);

  // Patch embedding: gather + WMMA GEMM
  { int total = B_*N_*768;
    gather_patches<<<(total+255)/256, blk, 0, stream>>>(x, f_xp, total); }
  gemm_wmma<<<dim3((E_+63)/64, (B_*N_+63)/64), blk, 0, stream>>>(
      f_xp, bf_patch, patch_b, nullptr, f_a, B_*N_, E_, 768, 0);

  // STRING (two circulant phase passes)
  dft_imag<<<2, 512, 0, stream>>>(circ, f_cim);
  string_pass<<<B_*N_, blk, 0, stream>>>(f_a, f_tmp, positions, f_cim, 0);
  string_pass<<<B_*N_, blk, 0, stream>>>(f_tmp, f_a, positions, f_cim, 1);

  // Prepend CLS token
  { int total = B_*S_*E_;
    concat_cls<<<(total+255)/256, blk, 0, stream>>>(f_a, cls_t, cls_p, f_h, total); }

  for (int l = 0; l < L_; ++l) {
    layernorm<<<B_*S_, blk, 0, stream>>>(f_h, f_a, ln1_g + l*E_, ln1_b + l*E_, B_*S_);
    gemm_wmma<<<dim3((3*E_+63)/64, (B_*S_+63)/64), blk, 0, stream>>>(
        f_a, bf_qkv + (size_t)l*3*E_*E_, bqkv + l*3*E_, nullptr, f_qkv, B_*S_, 3*E_, E_, 0);

    phi_q<<<B_*H_*S_, blk, 0, stream>>>(f_qkv, rf + (size_t)l*M_*DH_, f_qp);
    init_gmax<<<1, 1, 0, stream>>>(f_gmax);
    phi_k_logits<<<B_*H_*S_, blk, 0, stream>>>(f_qkv, rf + (size_t)l*M_*DH_, f_kp, f_gmax);
    { int total = B_*H_*S_*M_;
      phi_k_exp<<<(total+255)/256, blk, 0, stream>>>(f_kp, f_gmax, total); }

    kv_ksum<<<B_*H_, blk, 0, stream>>>(f_kp, f_qkv, f_kv, f_ks);
    num_den<<<B_*H_*S_, blk, 0, stream>>>(f_qp, f_kv, f_ks, f_a);

    // h += attn @ wo^T + bo  (residual fused in epilogue)
    gemm_wmma<<<dim3((E_+63)/64, (B_*S_+63)/64), blk, 0, stream>>>(
        f_a, bf_wo + (size_t)l*E_*E_, bo + l*E_, f_h, f_h, B_*S_, E_, E_, 0);

    layernorm<<<B_*S_, blk, 0, stream>>>(f_h, f_a, ln2_g + l*E_, ln2_b + l*E_, B_*S_);
    // mlp hidden with fused exact GELU
    gemm_wmma<<<dim3((MLP_+63)/64, (B_*S_+63)/64), blk, 0, stream>>>(
        f_a, bf_w1 + (size_t)l*MLP_*E_, b1 + l*MLP_, nullptr, f_kp, B_*S_, MLP_, E_, 1);
    // h += mlp @ w2^T + b2
    gemm_wmma<<<dim3((E_+63)/64, (B_*S_+63)/64), blk, 0, stream>>>(
        f_kp, bf_w2 + (size_t)l*E_*MLP_, b2 + l*E_, f_h, f_h, B_*S_, E_, MLP_, 0);
  }

  layernorm<<<B_*S_, blk, 0, stream>>>(f_h, f_a, norm_g, norm_b, B_*S_);
  { int total = B_*E_;
    extract_cls<<<(total+255)/256, blk, 0, stream>>>(f_a, f_cls, total); }
  gemm_wmma<<<dim3((NC_+63)/64, (B_+63)/64), blk, 0, stream>>>(
      f_cls, bf_head, head_b, nullptr, (float*)d_out, B_, NC_, E_, 0);
}
